// RelationalDelayGNNStage_15470472200856
// MI455X (gfx1250) — matrix operand order repaired and checked by reading, not compiled, not run on hardware
//
#include <hip/hip_runtime.h>
#include <hip/hip_bf16.h>

// ---------------------------------------------------------------------------
// RelationalDelayGNNStage for MI455X (gfx1250, wave32, WMMA)
//
// Per-node feature transform via bf16 WMMA GEMMs (fp32 accumulate) with the
// packed weight staged in LDS via CDNA5 async global->LDS copies, then masked
// edge gather + f32 atomic scatter (L2-resident working set), then residual +
// ReLU + L2-normalize per node.
// ---------------------------------------------------------------------------

#define NUM_LAYERS   4
#define N_EDGE_TYPES 3
#define DIM          128
#define N_WEIGHTS    13            // 3 edge + 10 hop
#define W_ELEMS      (DIM * DIM)   // 16384

typedef __attribute__((ext_vector_type(16))) __bf16 v16bf;
typedef __attribute__((ext_vector_type(8)))  __bf16 v8bf;
typedef __attribute__((ext_vector_type(8)))  float  v8f;

__device__ __forceinline__ unsigned short f2bf_rne(float f) {
    unsigned u = __builtin_bit_cast(unsigned, f);
    unsigned r = u + 0x7FFFu + ((u >> 16) & 1u);   // round-to-nearest-even
    return (unsigned short)(r >> 16);
}

// ---------------------------------------------------------------------------
// Pack W (fp32 [13][128][128], row = K, col = N) into the B-operand lane order
// of V_WMMA_F32_16X16X32_BF16 so the GEMM reads one contiguous 32B chunk/lane.
// Packed layout: Wp[w][kb][nt][lane][i], i=0..15:
//   K = kb*32 + (i>=8 ? 16 : 0) + (lane>=16 ? 8 : 0) + (i&7),  N = nt*16 + (lane&15)
// ---------------------------------------------------------------------------
__global__ __launch_bounds__(256) void pack_weights_kernel(
    const float* __restrict__ W_edge, const float* __restrict__ W_hop,
    unsigned short* __restrict__ Wp)
{
    int idx = blockIdx.x * 256 + threadIdx.x;
    if (idx >= N_WEIGHTS * W_ELEMS) return;
    int w = idx >> 14;            // /16384
    int p = idx & (W_ELEMS - 1);
    int i    = p & 15;
    int lane = (p >> 4) & 31;
    int nt   = (p >> 9) & 7;
    int kb   = p >> 12;
    int K = kb * 32 + (i >> 3) * 16 + ((lane >> 4) & 1) * 8 + (i & 7);
    int N = nt * 16 + (lane & 15);
    const float* Ws = (w < N_EDGE_TYPES) ? (W_edge + (size_t)w * W_ELEMS)
                                         : (W_hop + (size_t)(w - N_EDGE_TYPES) * W_ELEMS);
    Wp[idx] = f2bf_rne(Ws[(size_t)K * DIM + N]);
}

__global__ __launch_bounds__(256) void f32_to_bf16_kernel(
    const float* __restrict__ x, unsigned short* __restrict__ o, long n)
{
    long i = (long)blockIdx.x * 256 + threadIdx.x;
    if (i < n) o[i] = f2bf_rne(x[i]);
}

// acc[node][c] = sum of all biases used by this layer's sub-layers
__global__ __launch_bounds__(256) void init_acc_kernel(
    const float* __restrict__ b_edge, const float* __restrict__ b_hop,
    int hopBase, int hopCnt, float* __restrict__ acc, long total)
{
    long i = (long)blockIdx.x * 256 + threadIdx.x;
    if (i >= total) return;
    int c = (int)(i & (DIM - 1));
    float s = b_edge[c] + b_edge[DIM + c] + b_edge[2 * DIM + c];
    for (int j = 0; j < hopCnt; ++j) s += b_hop[(size_t)(hopBase + j) * DIM + c];
    acc[i] = s;
}

// ---------------------------------------------------------------------------
// H[N,128] (fp32) = A[N,128] (bf16) @ Wp (bf16 packed), via
// v_wmma_f32_16x16x32_bf16. One wave per 16-row strip; 8 N-tiles x 4 K-blocks.
// The 32KB packed W is staged into LDS once per workgroup with CDNA5 async
// global->LDS copies (global_load_async_to_lds_b128 / s_wait_asynccnt), so
// all 8 waves feed their B fragments from ds_load_b128 instead of re-reading
// 32KB each from L2.
// A fragment layout (ISA "16-bit A-Matrix 16x32"):
//   lane m=lane&15 reads row m; elems 0..7: K = kb*32 + (lane>=16?8:0) + e
//                              elems 8..15: K = that + 16.
// C layout: vgpr r -> row = r + (lane>=16?8:0), col = nt*16 + (lane&15).
// ---------------------------------------------------------------------------
__global__ __launch_bounds__(256) void gemm_bf16_wmma_kernel(
    const unsigned short* __restrict__ Abf,
    const unsigned short* __restrict__ Wp,
    float* __restrict__ H, int nRows)
{
    __shared__ unsigned short sW[W_ELEMS];   // 32 KB packed B fragments

    const int tid = threadIdx.x;

    // ---- cooperative async stage of W into LDS (2048 x 16B chunks) ----
#pragma unroll
    for (int it = 0; it < 8; ++it) {
        const unsigned goff = (unsigned)((it * 256 + tid) * 16);   // byte offset
        const unsigned ldsAddr =
            (unsigned)(size_t)(const void*)((const char*)sW + goff);
        asm volatile("global_load_async_to_lds_b128 %0, %1, %2"
                     :: "v"(ldsAddr), "v"(goff), "s"(Wp)
                     : "memory");
    }
    asm volatile("s_wait_asynccnt 0x0" ::: "memory");
    __syncthreads();

    const int lane  = tid & 31;
    const int wave  = tid >> 5;
    const int strip = blockIdx.x * 8 + wave;
    if (strip * 16 >= nRows) return;               // wave-uniform: EXEC stays full
    const int row0 = strip * 16;
    const int m  = lane & 15;
    const int hi = (lane >> 4) & 1;

    const unsigned short* arow = Abf + (size_t)(row0 + m) * DIM;
    v16bf a[4];
#pragma unroll
    for (int kb = 0; kb < 4; ++kb) {
        const int c0 = kb * 32 + hi * 8;
        v8bf lo  = *reinterpret_cast<const v8bf*>(arow + c0);
        v8bf hi8 = *reinterpret_cast<const v8bf*>(arow + c0 + 16);
        a[kb] = __builtin_shufflevector(lo, hi8, 0,1,2,3,4,5,6,7,8,9,10,11,12,13,14,15);
    }

#pragma unroll
    for (int nt = 0; nt < 8; ++nt) {
        v8f c = {};
#pragma unroll
        for (int kb = 0; kb < 4; ++kb) {
            const unsigned short* bp = sW + (size_t)(((kb * 8 + nt) * 32 + lane) * 16);
            v8bf b0 = *reinterpret_cast<const v8bf*>(bp);
            v8bf b1 = *reinterpret_cast<const v8bf*>(bp + 8);
            v16bf b = __builtin_shufflevector(b0, b1, 0,1,2,3,4,5,6,7,8,9,10,11,12,13,14,15);
            c = __builtin_amdgcn_wmma_f32_16x16x32_bf16(
                    false, a[kb], false, b, (short)0, c, false, false);
        }
        float* crow = H + (size_t)(row0 + hi * 8) * DIM + nt * 16 + m;
#pragma unroll
        for (int r = 0; r < 8; ++r) crow[(size_t)r * DIM] = c[r];
    }
}

// ---------------------------------------------------------------------------
// One wave per edge; wave-uniform mask test; lane moves float4 with 4 relaxed
// agent-scope f32 atomic adds (-> global_atomic_add_f32, resolves in L2).
// ---------------------------------------------------------------------------
__global__ __launch_bounds__(256) void scatter_edges_kernel(
    const float* __restrict__ H, const int* __restrict__ src,
    const int* __restrict__ dst, const int* __restrict__ sel, int target,
    float* __restrict__ acc, int nEdges)
{
    const int lane = threadIdx.x & 31;
    const int e = blockIdx.x * 8 + (threadIdx.x >> 5);
    if (e >= nEdges) return;
    if (sel[e] != target) return;
    const int s = src[e], d = dst[e];
    const float4 v = *reinterpret_cast<const float4*>(H + (size_t)s * DIM + lane * 4);
    float* ap = acc + (size_t)d * DIM + lane * 4;
    __hip_atomic_fetch_add(ap + 0, v.x, __ATOMIC_RELAXED, __HIP_MEMORY_SCOPE_AGENT);
    __hip_atomic_fetch_add(ap + 1, v.y, __ATOMIC_RELAXED, __HIP_MEMORY_SCOPE_AGENT);
    __hip_atomic_fetch_add(ap + 2, v.z, __ATOMIC_RELAXED, __HIP_MEMORY_SCOPE_AGENT);
    __hip_atomic_fetch_add(ap + 3, v.w, __ATOMIC_RELAXED, __HIP_MEMORY_SCOPE_AGENT);
}

// ---------------------------------------------------------------------------
// cur = (x + relu(acc)) / max(||.||_2, eps); wave per node, shfl_xor reduction.
// Writes fp32 next state (or d_out) and optional bf16 copy for next GEMMs.
// ---------------------------------------------------------------------------
__global__ __launch_bounds__(256) void combine_norm_kernel(
    const float* __restrict__ xin, const float* __restrict__ acc,
    float* __restrict__ outF, unsigned short* __restrict__ outB, int nNodes)
{
    const int lane = threadIdx.x & 31;
    const int node = blockIdx.x * 8 + (threadIdx.x >> 5);
    if (node >= nNodes) return;
    const size_t base = (size_t)node * DIM + lane * 4;
    const float4 xv = *reinterpret_cast<const float4*>(xin + base);
    const float4 av = *reinterpret_cast<const float4*>(acc + base);
    float4 v;
    v.x = xv.x + fmaxf(av.x, 0.f);
    v.y = xv.y + fmaxf(av.y, 0.f);
    v.z = xv.z + fmaxf(av.z, 0.f);
    v.w = xv.w + fmaxf(av.w, 0.f);
    float ss = v.x * v.x + v.y * v.y + v.z * v.z + v.w * v.w;
#pragma unroll
    for (int msk = 16; msk; msk >>= 1) ss += __shfl_xor(ss, msk, 32);
    const float scale = 1.0f / fmaxf(sqrtf(ss), 1e-12f);
    v.x *= scale; v.y *= scale; v.z *= scale; v.w *= scale;
    *reinterpret_cast<float4*>(outF + base) = v;
    if (outB) {
        uint2 p;
        p.x = (unsigned)f2bf_rne(v.x) | ((unsigned)f2bf_rne(v.y) << 16);
        p.y = (unsigned)f2bf_rne(v.z) | ((unsigned)f2bf_rne(v.w) << 16);
        *reinterpret_cast<uint2*>(outB + base) = p;
    }
}

// ---------------------------------------------------------------------------
extern "C" void kernel_launch(void* const* d_in, const int* in_sizes, int n_in,
                              void* d_out, int out_size, void* d_ws, size_t ws_size,
                              hipStream_t stream)
{
    const float* x        = (const float*)d_in[0];
    const int*   ei       = (const int*)d_in[1];   // [2, E]
    const int*   ehop     = (const int*)d_in[2];   // [E]
    const int*   etype    = (const int*)d_in[3];   // [E]
    const float* W_edge   = (const float*)d_in[4];
    const float* b_edge   = (const float*)d_in[5];
    const float* W_hop    = (const float*)d_in[6];
    const float* b_hop    = (const float*)d_in[7];

    const int N = in_sizes[0] / DIM;               // 50000
    const int E = in_sizes[2];                     // 600000
    const int* srcIdx = ei;
    const int* dstIdx = ei + E;
    const long NB = (long)N * DIM;                 // elements per state

    // ---- workspace carve-up (all 256B aligned since NB*4 is) ----
    char* w = (char*)d_ws;
    float* stateF = (float*)w;            w += (size_t)3 * NB * sizeof(float);   // xs[1..3] fp32
    unsigned short* stateB = (unsigned short*)w; w += (size_t)4 * NB * sizeof(unsigned short); // xs[0..3] bf16
    float* accBuf = (float*)w;            w += (size_t)NB * sizeof(float);
    float* Hbuf   = (float*)w;            w += (size_t)NB * sizeof(float);
    unsigned short* Wp = (unsigned short*)w;  // 13 * 16384 bf16

    const float* xsF[NUM_LAYERS];
    unsigned short* xsB[NUM_LAYERS];
    xsF[0] = x;
    for (int t = 1; t < NUM_LAYERS; ++t) xsF[t] = stateF + (size_t)(t - 1) * NB;
    for (int t = 0; t < NUM_LAYERS; ++t) xsB[t] = stateB + (size_t)t * NB;

    // ---- one-time prep ----
    {
        int total = N_WEIGHTS * W_ELEMS;
        pack_weights_kernel<<<(total + 255) / 256, 256, 0, stream>>>(W_edge, W_hop, Wp);
        f32_to_bf16_kernel<<<(int)((NB + 255) / 256), 256, 0, stream>>>(x, xsB[0], NB);
    }

    const int strips      = (N + 15) / 16;
    const int gemmBlocks  = (strips + 7) / 8;
    const int edgeBlocks  = (E + 7) / 8;
    const int nodeBlocks  = (N + 7) / 8;
    const int elemBlocks  = (int)((NB + 255) / 256);

    for (int t = 0; t < NUM_LAYERS; ++t) {
        const int hopBase = t * (t + 1) / 2;
        const int hopCnt  = t + 1;

        init_acc_kernel<<<elemBlocks, 256, 0, stream>>>(
            b_edge, b_hop, hopBase, hopCnt, accBuf, NB);

        // relation-specific sub-layers on xs[t]
        for (int e = 0; e < N_EDGE_TYPES; ++e) {
            gemm_bf16_wmma_kernel<<<gemmBlocks, 256, 0, stream>>>(
                xsB[t], Wp + (size_t)e * W_ELEMS, Hbuf, N);
            scatter_edges_kernel<<<edgeBlocks, 256, 0, stream>>>(
                Hbuf, srcIdx, dstIdx, etype, e, accBuf, E);
        }
        // k-hop sub-layers: k = 1..t+1 read delayed state xs[t-(k-1)]
        for (int k = 1; k <= t + 1; ++k) {
            const int widx = N_EDGE_TYPES + hopBase + (k - 1);
            const int sIdx = t - (k - 1);
            gemm_bf16_wmma_kernel<<<gemmBlocks, 256, 0, stream>>>(
                xsB[sIdx], Wp + (size_t)widx * W_ELEMS, Hbuf, N);
            scatter_edges_kernel<<<edgeBlocks, 256, 0, stream>>>(
                Hbuf, srcIdx, dstIdx, ehop, k, accBuf, E);
        }

        float* outF = (t + 1 < NUM_LAYERS) ? (float*)(stateF + (size_t)t * NB) : (float*)d_out;
        unsigned short* outB = (t + 1 < NUM_LAYERS) ? xsB[t + 1] : nullptr;
        combine_norm_kernel<<<nodeBlocks, 256, 0, stream>>>(
            xsF[t], accBuf, outF, outB, N);
    }
}
